// MambaBlock_14267881357638
// MI455X (gfx1250) — matrix operand order repaired and checked by reading, not compiled, not run on hardware
//
#include <hip/hip_runtime.h>
#include <hip/hip_bf16.h>
#include <math.h>

typedef __attribute__((ext_vector_type(16))) _Float16 v16h;
typedef __attribute__((ext_vector_type(8)))  _Float16 v8h;
typedef __attribute__((ext_vector_type(8)))  float    v8f;
typedef __attribute__((ext_vector_type(4)))  unsigned v4u;
typedef __attribute__((ext_vector_type(8)))  int      v8i;
typedef __attribute__((ext_vector_type(4)))  int      v4i;

#define L_SEQ   1024
#define D_MODEL 1536
#define D_INNER 3072
#define DT_RANK 96
#define D_STATE 16
#define D_CONV  4
#define E2      (2*D_INNER)            // 6144
#define XDBLW   (DT_RANK + 2*D_STATE)  // 128

#define BM 64
#define BN 128
#define BK 32
#define NCH 32                          // scan chunks
#define CL  (L_SEQ / NCH)               // 32 steps per chunk

#if defined(__has_builtin)
#if __has_builtin(__builtin_amdgcn_tensor_load_to_lds) && __has_builtin(__builtin_amdgcn_s_wait_tensorcnt)
#define HAVE_TDM 1
#endif
#endif
#ifndef HAVE_TDM
#define HAVE_TDM 0
#endif

// ---------------- fp32 -> fp16 convert (supports row slicing via src_ld) ----------------
__global__ void k_cvt_f16(const float* __restrict__ src, _Float16* __restrict__ dst,
                          int rows, int cols, int src_ld) {
    int i = blockIdx.x * blockDim.x + threadIdx.x;
    if (i >= rows * cols) return;
    int r = i / cols, c = i - r * cols;
    dst[i] = (_Float16)src[(size_t)r * src_ld + c];
}

// tiled transpose convert: src [rows][cols] f32 -> dst [cols][rows] f16 (coalesced both sides)
__global__ __launch_bounds__(256)
void k_cvt_f16_T(const float* __restrict__ src, _Float16* __restrict__ dst,
                 int rows, int cols) {
    __shared__ float tile[32][33];                   // +1 pad: no bank conflicts
    const int bx = blockIdx.x * 32;                  // col base
    const int by = blockIdx.y * 32;                  // row base
    const int tx = threadIdx.x & 31;
    const int ty = threadIdx.x >> 5;                 // 0..7
    #pragma unroll
    for (int j = 0; j < 32; j += 8)
        tile[ty + j][tx] = src[(size_t)(by + ty + j) * cols + bx + tx];
    __syncthreads();
    #pragma unroll
    for (int j = 0; j < 32; j += 8)
        dst[(size_t)(bx + ty + j) * rows + by + tx] = (_Float16)tile[tx][ty + j];
}

// ---------------- CDNA5 async global->LDS DMA (ASYNCcnt-tracked) ----------------
__device__ __forceinline__ void async_b128(unsigned lds_off, const void* gaddr) {
    asm volatile("global_load_async_to_lds_b128 %0, %1, off"
                 :: "v"(lds_off), "v"(gaddr) : "memory");
}

#if HAVE_TDM
// ---------------- CDNA5 Tensor Data Mover: 2D tile [BM rows x BK cols] f16 -> LDS --------
__device__ __forceinline__ void tdm_load_A(unsigned ldsOff, unsigned long long gaddr, int lda) {
    v4u g0;
    g0[0] = 1u;                                              // count=1, user desc, no gather
    g0[1] = ldsOff;                                          // lds_addr (bytes)
    g0[2] = (unsigned)gaddr;                                 // global_addr[31:0]
    g0[3] = (unsigned)((gaddr >> 32) & 0x01FFFFFFu) | (2u << 30); // addr[56:32] | type=2
    v8i g1;
    g1[0] = (int)(1u << 16);                                 // data_size=1 (2 bytes), mask=0
    g1[1] = (int)(((unsigned)lda & 0xFFFFu) << 16);          // tensor_dim0[15:0]
    g1[2] = (int)((((unsigned)lda >> 16) & 0xFFFFu) | ((unsigned)BM << 16)); // dim0 hi | dim1 lo
    g1[3] = (int)((unsigned)BK << 16);                       // dim1 hi=0 | tile_dim0=BK
    g1[4] = BM;                                              // tile_dim1=BM, tile_dim2=0
    g1[5] = lda;                                             // tensor_dim0_stride[31:0]
    g1[6] = 0;
    g1[7] = 0;
    v4i gz4 = {0, 0, 0, 0};
    v8i gz8 = {0, 0, 0, 0, 0, 0, 0, 0};
    // 6-arg toolchain variant: (g0, g1, g2, g3, extra_group, cpol)
    __builtin_amdgcn_tensor_load_to_lds(g0, g1, gz4, gz4, gz8, 0);
}
#endif

// A fragment 16x32 f16 from row-major LDS subtile (ld = 32 halves)
__device__ __forceinline__ v16h fragA(const _Float16* base, int r16, int lhalf) {
    v8h lo = *(const v8h*)(base + r16 * BK + lhalf * 8);
    v8h hi = *(const v8h*)(base + r16 * BK + 16 + lhalf * 8);
    v16h a;
    #pragma unroll
    for (int i = 0; i < 8; ++i) { a[i] = lo[i]; a[i + 8] = hi[i]; }
    return a;
}
// B fragment 32x16 f16: N = r16 (LDS row), lane<16 holds K 0..7,16..23, lane>=16 K 8..15,24..31
__device__ __forceinline__ v16h fragB(const _Float16* base, int r16, int lhalf) {
    v8h lo = *(const v8h*)(base + r16 * BK + lhalf * 8);
    v8h hi = *(const v8h*)(base + r16 * BK + 16 + lhalf * 8);
    v16h b;
    #pragma unroll
    for (int i = 0; i < 8; ++i) { b[i] = lo[i]; b[i + 8] = hi[i]; }
    return b;
}

__device__ __forceinline__ void storeC(float* C, int ldc, int tm, int tn,
                                       const v8f& acc, int r16, int lhalf, int storeCT) {
    const int moff = lhalf * 8;
    if (storeCT) {
        float* cp = C + (size_t)(tn + r16) * ldc + tm + moff;
        #pragma unroll
        for (int v = 0; v < 8; ++v) cp[v] = acc[v];
    } else {
        #pragma unroll
        for (int v = 0; v < 8; ++v)
            C[(size_t)(tm + moff + v) * ldc + tn + r16] = acc[v];
    }
}

// ---- WMMA GEMM: C[M][N] (or C[N][M] if storeCT) = A[M][K] * B[N][K]^T ----
// grid = (N/BN, M/BM), block 256 = 8 waves in 2(M)x4(N); each wave: 32x32 out (4 WMMA/step)
// A panel: TDM tensor_load_to_lds (wave 0, TENSORcnt); B panel: per-lane async b128 (ASYNCcnt)
__global__ __launch_bounds__(256)
void k_gemm_f16(const _Float16* __restrict__ A, int lda,
                const _Float16* __restrict__ B, int ldb,
                float* __restrict__ C, int ldc,
                int K, int storeCT) {
    __shared__ __align__(16) _Float16 sA[2][BM * BK];   // 2 x 4 KB
    __shared__ __align__(16) _Float16 sB[2][BN * BK];   // 2 x 8 KB

    const int tid   = threadIdx.x;
    const int lane  = tid & 31;
    const int wave  = tid >> 5;
    const int wm    = (wave >> 2) * 32;     // wave M offset in block tile
    const int wn    = (wave & 3) * 32;      // wave N offset in block tile
    const int m0    = blockIdx.y * BM;
    const int n0    = blockIdx.x * BN;
    const int lhalf = lane >> 4;
    const int r16   = lane & 15;

    // B async copy: thread moves two b128 per stage
    const int row  = tid >> 2;              // 0..63
    const int cele = (tid & 3) * 8;         // element offset 0,8,16,24 (16B aligned)
    const _Float16* gB0 = B + (size_t)(n0 + row) * ldb + cele;
    const _Float16* gB1 = B + (size_t)(n0 + 64 + row) * ldb + cele;
    unsigned oB0[2], oB1[2];
    #pragma unroll
    for (int b = 0; b < 2; ++b) {
        oB0[b] = (unsigned)(size_t)&sB[b][row * BK + cele];
        oB1[b] = (unsigned)(size_t)&sB[b][(64 + row) * BK + cele];
    }
#if HAVE_TDM
    const unsigned long long gAbase = (unsigned long long)(size_t)(A + (size_t)m0 * lda);
    const unsigned oA[2] = { (unsigned)(size_t)&sA[0][0], (unsigned)(size_t)&sA[1][0] };
#else
    const _Float16* gA = A + (size_t)(m0 + row) * lda + cele;
    unsigned oA[2];
    #pragma unroll
    for (int b = 0; b < 2; ++b) oA[b] = (unsigned)(size_t)&sA[b][row * BK + cele];
#endif

    v8f a00 = {}, a01 = {}, a10 = {}, a11 = {};

    const int nstage = K / BK;
    // prologue: stage 0 into buffer 0
#if HAVE_TDM
    if (wave == 0) tdm_load_A(oA[0], gAbase, lda);
#else
    async_b128(oA[0], gA);
#endif
    async_b128(oB0[0], gB0);
    async_b128(oB1[0], gB1);

    for (int s = 0; s < nstage; ++s) {
        const int cur = s & 1;
        if (s + 1 < nstage) {
            const int k1 = (s + 1) * BK;
#if HAVE_TDM
            if (wave == 0) tdm_load_A(oA[cur ^ 1], gAbase + (size_t)k1 * 2, lda);
#else
            async_b128(oA[cur ^ 1], gA + k1);
#endif
            async_b128(oB0[cur ^ 1], gB0 + k1);
            async_b128(oB1[cur ^ 1], gB1 + k1);
            __builtin_prefetch(gB0 + k1 + BK, 0, 3);
#if HAVE_TDM
            if (wave == 0) __builtin_amdgcn_s_wait_tensorcnt((short)1);
            asm volatile("s_wait_asynccnt 0x2" ::: "memory");   // stage s done, s+1 in flight
#else
            asm volatile("s_wait_asynccnt 0x3" ::: "memory");
#endif
        } else {
#if HAVE_TDM
            if (wave == 0) __builtin_amdgcn_s_wait_tensorcnt((short)0);
#endif
            asm volatile("s_wait_asynccnt 0x0" ::: "memory");
        }
        __syncthreads();

        v16h af0 = fragA(&sA[cur][(wm)      * BK], r16, lhalf);
        v16h af1 = fragA(&sA[cur][(wm + 16) * BK], r16, lhalf);
        v16h bf0 = fragB(&sB[cur][(wn)      * BK], r16, lhalf);
        v16h bf1 = fragB(&sB[cur][(wn + 16) * BK], r16, lhalf);

        a00 = __builtin_amdgcn_wmma_f32_16x16x32_f16(false, af0, false, bf0, (short)0, a00, false, false);
        a01 = __builtin_amdgcn_wmma_f32_16x16x32_f16(false, af0, false, bf1, (short)0, a01, false, false);
        a10 = __builtin_amdgcn_wmma_f32_16x16x32_f16(false, af1, false, bf0, (short)0, a10, false, false);
        a11 = __builtin_amdgcn_wmma_f32_16x16x32_f16(false, af1, false, bf1, (short)0, a11, false, false);

        __syncthreads();
    }

    storeC(C, ldc, m0 + wm,      n0 + wn,      a00, r16, lhalf, storeCT);
    storeC(C, ldc, m0 + wm,      n0 + wn + 16, a01, r16, lhalf, storeCT);
    storeC(C, ldc, m0 + wm + 16, n0 + wn,      a10, r16, lhalf, storeCT);
    storeC(C, ldc, m0 + wm + 16, n0 + wn + 16, a11, r16, lhalf, storeCT);
}

// ---------------- depthwise causal conv (k=4) + bias + SiLU; x,xc in [d][l] ---------------
__global__ void k_conv_silu(const float* __restrict__ x, const float* __restrict__ cw,
                            const float* __restrict__ cb, float* __restrict__ xc) {
    int i = blockIdx.x * blockDim.x + threadIdx.x;
    if (i >= D_INNER * L_SEQ) return;
    int d = i / L_SEQ, l = i - d * L_SEQ;
    float acc = cb[d];
    #pragma unroll
    for (int j = 0; j < D_CONV; ++j) {
        int li = l - (D_CONV - 1) + j;
        if (li >= 0) acc += cw[d * D_CONV + j] * x[(size_t)d * L_SEQ + li];
    }
    xc[i] = acc / (1.f + __expf(-acc));   // silu
}

__global__ void k_conv_state(const float* __restrict__ x, float* __restrict__ cs) {
    int i = blockIdx.x * blockDim.x + threadIdx.x;
    if (i >= D_INNER * D_CONV) return;
    int d = i / D_CONV, j = i - d * D_CONV;
    cs[i] = x[(size_t)d * L_SEQ + (L_SEQ - D_CONV) + j];
}

// ---------------- chunked parallel scan ----------------
__device__ __forceinline__ float softplusf(float x) {
    return (x > 20.f) ? x : log1pf(__expf(x));
}

// Phase A: per (d, chunk) compute (P = prod dA, U = h assuming h_in = 0)
__global__ void k_scan_chunk(const float* __restrict__ dpre, const float* __restrict__ dt_b,
                             const float* __restrict__ xdbl, const float* __restrict__ xc,
                             const float* __restrict__ A_log,
                             float* __restrict__ Pbuf, float* __restrict__ Ubuf) {
    int i = blockIdx.x * blockDim.x + threadIdx.x;
    if (i >= D_INNER * NCH) return;
    int d = i / NCH, c = i - d * NCH;
    float negA[D_STATE], P[D_STATE], U[D_STATE];
    #pragma unroll
    for (int n = 0; n < D_STATE; ++n) {
        negA[n] = -__expf(A_log[(size_t)d * D_STATE + n]);
        P[n] = 1.f; U[n] = 0.f;
    }
    const float db = dt_b[d];
    const int l0 = c * CL;
    for (int l = l0; l < l0 + CL; ++l) {
        float delta = softplusf(dpre[(size_t)d * L_SEQ + l] + db);
        float dx = delta * xc[(size_t)d * L_SEQ + l];
        const float* bc = xdbl + (size_t)l * XDBLW + DT_RANK;
        #pragma unroll
        for (int n = 0; n < D_STATE; ++n) {
            float dA = __expf(delta * negA[n]);
            U[n] = dA * U[n] + dx * bc[n];
            P[n] *= dA;
        }
    }
    #pragma unroll
    for (int n = 0; n < D_STATE; ++n) {
        Pbuf[(size_t)i * D_STATE + n] = P[n];
        Ubuf[(size_t)i * D_STATE + n] = U[n];
    }
}

// Phase B: prefix over chunk summaries, one thread per (d, n)
__global__ void k_scan_prefix(const float* __restrict__ Pbuf, const float* __restrict__ Ubuf,
                              float* __restrict__ Hin) {
    int i = blockIdx.x * blockDim.x + threadIdx.x;
    if (i >= D_INNER * D_STATE) return;
    int d = i / D_STATE, n = i - d * D_STATE;
    float h = 0.f;
    for (int c = 0; c < NCH; ++c) {
        size_t idx = ((size_t)d * NCH + c) * D_STATE + n;
        Hin[idx] = h;
        h = Pbuf[idx] * h + Ubuf[idx];
    }
}

// Phase C: replay chunk with correct incoming state; fuse C-proj, D-skip, silu(z) gate
__global__ void k_scan_apply(const float* __restrict__ dpre, const float* __restrict__ dt_b,
                             const float* __restrict__ xdbl, const float* __restrict__ xc,
                             const float* __restrict__ z, const float* __restrict__ A_log,
                             const float* __restrict__ Dv, const float* __restrict__ Hin,
                             float* __restrict__ yz, float* __restrict__ last_state) {
    int i = blockIdx.x * blockDim.x + threadIdx.x;
    if (i >= D_INNER * NCH) return;
    int d = i / NCH, c = i - d * NCH;
    float negA[D_STATE], h[D_STATE];
    #pragma unroll
    for (int n = 0; n < D_STATE; ++n) {
        negA[n] = -__expf(A_log[(size_t)d * D_STATE + n]);
        h[n] = Hin[(size_t)i * D_STATE + n];
    }
    const float db = dt_b[d];
    const float Dd = Dv[d];
    const int l0 = c * CL;
    for (int l = l0; l < l0 + CL; ++l) {
        float delta = softplusf(dpre[(size_t)d * L_SEQ + l] + db);
        float xcv = xc[(size_t)d * L_SEQ + l];
        float dx = delta * xcv;
        const float* bc = xdbl + (size_t)l * XDBLW + DT_RANK;
        float yacc = 0.f;
        #pragma unroll
        for (int n = 0; n < D_STATE; ++n) {
            float dA = __expf(delta * negA[n]);
            h[n] = dA * h[n] + dx * bc[n];
            yacc += h[n] * bc[D_STATE + n];
        }
        float y = yacc + Dd * xcv;
        float zv = z[(size_t)d * L_SEQ + l];
        yz[(size_t)d * L_SEQ + l] = y * (zv / (1.f + __expf(-zv)));
    }
    if (c == NCH - 1) {
        #pragma unroll
        for (int n = 0; n < D_STATE; ++n)
            last_state[(size_t)d * D_STATE + n] = h[n];
    }
}

// =========================================================================================
extern "C" void kernel_launch(void* const* d_in, const int* in_sizes, int n_in,
                              void* d_out, int out_size, void* d_ws, size_t ws_size,
                              hipStream_t stream) {
    const float* hs    = (const float*)d_in[0];
    const float* w_in  = (const float*)d_in[1];
    const float* cw    = (const float*)d_in[2];
    const float* cb    = (const float*)d_in[3];
    const float* w_xp  = (const float*)d_in[4];
    const float* w_dt  = (const float*)d_in[5];
    const float* b_dt  = (const float*)d_in[6];
    const float* w_out = (const float*)d_in[7];
    const float* A_log = (const float*)d_in[8];
    const float* Dv    = (const float*)d_in[9];

    float* out_main = (float*)d_out;                                   // [1024][1536]
    float* out_cs   = out_main + (size_t)L_SEQ * D_MODEL;              // [3072][4]
    float* out_ls   = out_cs + (size_t)D_INNER * D_CONV;               // [3072][16]

    char* p = (char*)d_ws;
    auto alloc = [&](size_t bytes) -> void* {
        void* q = (void*)p;
        p += (bytes + 255) & ~(size_t)255;
        return q;
    };
    _Float16* hsF16  = (_Float16*)alloc((size_t)L_SEQ  * D_MODEL * 2);
    _Float16* winF16 = (_Float16*)alloc((size_t)E2     * D_MODEL * 2);
    _Float16* xpF16  = (_Float16*)alloc((size_t)XDBLW  * D_INNER * 2);
    _Float16* dtwF16 = (_Float16*)alloc((size_t)D_INNER* DT_RANK * 2);
    _Float16* woF16  = (_Float16*)alloc((size_t)D_MODEL* D_INNER * 2);
    float*    xzT    = (float*)   alloc((size_t)E2     * L_SEQ   * 4);   // [e][l]
    float*    xcbuf  = (float*)   alloc((size_t)D_INNER* L_SEQ   * 4);   // [d][l]
    _Float16* xcT16  = (_Float16*)alloc((size_t)L_SEQ  * D_INNER * 2);   // [l][d]
    float*    xdbl   = (float*)   alloc((size_t)L_SEQ  * XDBLW   * 4);   // [l][128]
    _Float16* dt16   = (_Float16*)alloc((size_t)L_SEQ  * DT_RANK * 2);   // [l][96]
    float*    dpre   = (float*)   alloc((size_t)D_INNER* L_SEQ   * 4);   // [d][l]
    float*    yzbuf  = (float*)   alloc((size_t)D_INNER* L_SEQ   * 4);   // [d][l]
    _Float16* y16T   = (_Float16*)alloc((size_t)L_SEQ  * D_INNER * 2);   // [l][d]
    float*    Pbuf   = (float*)   alloc((size_t)D_INNER* NCH * D_STATE * 4);
    float*    Ubuf   = (float*)   alloc((size_t)D_INNER* NCH * D_STATE * 4);
    float*    Hin    = (float*)   alloc((size_t)D_INNER* NCH * D_STATE * 4);

    float* xbuf = xzT;                                 // first D_INNER rows [d][l]
    float* zbuf = xzT + (size_t)D_INNER * L_SEQ;       // next D_INNER rows

    auto g1 = [](size_t n) { return (unsigned)((n + 255) / 256); };

    // --- fp32 -> fp16 conversions ---
    k_cvt_f16<<<g1((size_t)L_SEQ * D_MODEL), 256, 0, stream>>>(hs,    hsF16,  L_SEQ,  D_MODEL, D_MODEL);
    k_cvt_f16<<<g1((size_t)E2 * D_MODEL),    256, 0, stream>>>(w_in,  winF16, E2,     D_MODEL, D_MODEL);
    k_cvt_f16<<<g1((size_t)XDBLW * D_INNER), 256, 0, stream>>>(w_xp,  xpF16,  XDBLW,  D_INNER, D_INNER);
    k_cvt_f16<<<g1((size_t)D_INNER * DT_RANK),256,0, stream>>>(w_dt,  dtwF16, D_INNER,DT_RANK, DT_RANK);
    k_cvt_f16<<<g1((size_t)D_MODEL * D_INNER),256,0, stream>>>(w_out, woF16,  D_MODEL,D_INNER, D_INNER);

    // --- GEMM 1: xz^T [6144][1024] = (hs[1024][1536] * W_in[6144][1536]^T)^T ---
    k_gemm_f16<<<dim3(E2 / BN, L_SEQ / BM), 256, 0, stream>>>(
        hsF16, D_MODEL, winF16, D_MODEL, xzT, L_SEQ, D_MODEL, /*storeCT=*/1);

    // --- depthwise conv + SiLU, conv_state ---
    k_conv_silu <<<g1((size_t)D_INNER * L_SEQ),  256, 0, stream>>>(xbuf, cw, cb, xcbuf);
    k_conv_state<<<g1((size_t)D_INNER * D_CONV), 256, 0, stream>>>(xbuf, out_cs);

    // --- GEMM 2: x_dbl [1024][128] = xc^T[1024][3072] * W_xp[128][3072]^T ---
    k_cvt_f16_T<<<dim3(L_SEQ / 32, D_INNER / 32), 256, 0, stream>>>(xcbuf, xcT16, D_INNER, L_SEQ);
    k_gemm_f16<<<dim3(XDBLW / BN, L_SEQ / BM), 256, 0, stream>>>(
        xcT16, D_INNER, xpF16, D_INNER, xdbl, XDBLW, D_INNER, /*storeCT=*/0);

    // --- GEMM 3: delta_pre^T [3072][1024] = (dt[1024][96] * W_dt[3072][96]^T)^T ---
    k_cvt_f16<<<g1((size_t)L_SEQ * DT_RANK), 256, 0, stream>>>(xdbl, dt16, L_SEQ, DT_RANK, XDBLW);
    k_gemm_f16<<<dim3(D_INNER / BN, L_SEQ / BM), 256, 0, stream>>>(
        dt16, DT_RANK, dtwF16, DT_RANK, dpre, L_SEQ, DT_RANK, /*storeCT=*/1);

    // --- chunked parallel scan: A (chunk summaries), B (prefix), C (apply + gate) ---
    k_scan_chunk <<<g1((size_t)D_INNER * NCH),     256, 0, stream>>>(dpre, b_dt, xdbl, xcbuf, A_log, Pbuf, Ubuf);
    k_scan_prefix<<<g1((size_t)D_INNER * D_STATE), 256, 0, stream>>>(Pbuf, Ubuf, Hin);
    k_scan_apply <<<g1((size_t)D_INNER * NCH),     256, 0, stream>>>(dpre, b_dt, xdbl, xcbuf, zbuf, A_log,
                                                                     Dv, Hin, yzbuf, out_ls);

    // --- GEMM 4: out [1024][1536] = yz^T[1024][3072] * W_out[1536][3072]^T ---
    k_cvt_f16_T<<<dim3(L_SEQ / 32, D_INNER / 32), 256, 0, stream>>>(yzbuf, y16T, D_INNER, L_SEQ);
    k_gemm_f16<<<dim3(D_MODEL / BN, L_SEQ / BM), 256, 0, stream>>>(
        y16T, D_INNER, woF16, D_INNER, out_main, D_MODEL, D_INNER, /*storeCT=*/0);

    (void)in_sizes; (void)n_in; (void)out_size; (void)ws_size;
}
